// MambaBlock_29721173688724
// MI455X (gfx1250) — matrix operand order repaired and checked by reading, not compiled, not run on hardware
//
#include <hip/hip_runtime.h>

#define DIM      1024
#define D_STATE  16
#define D_CONV   4
#define D_INNER  2048
#define DT_RANK  64
#define SEQ      2048
#define BATCH    2
#define NROWS    (BATCH * SEQ)            // 4096
#define DBC_W    (DT_RANK + 2 * D_STATE)  // 96

typedef __attribute__((ext_vector_type(16))) __bf16        v16bf;
typedef __attribute__((ext_vector_type(8)))  float         v8f;
typedef __attribute__((ext_vector_type(4)))  unsigned int  u32x4;

union FragAB { v16bf v; u32x4 q[2]; };

__device__ __forceinline__ unsigned short f2bf(float f) {
  unsigned int u = __float_as_uint(f);
  unsigned int r = u + 0x7FFFu + ((u >> 16) & 1u);   // round-to-nearest-even
  return (unsigned short)(r >> 16);
}
__device__ __forceinline__ float bf2f(unsigned short h) {
  return __uint_as_float(((unsigned int)h) << 16);
}

// ---------------------------------------------------------------- cvt weights
__global__ void cvt_f32_to_bf16(const float* __restrict__ src,
                                unsigned short* __restrict__ dst, int n) {
  int i = blockIdx.x * blockDim.x + threadIdx.x;
  if (i < n) dst[i] = f2bf(src[i]);
}

// ---------------------------------------------------------------- layernorm -> bf16
__global__ void layernorm_bf16_kernel(const float* __restrict__ x,
                                      const float* __restrict__ w,
                                      const float* __restrict__ b,
                                      unsigned short* __restrict__ out) {
  __shared__ float red_s[8];
  __shared__ float red_ss[8];
  int row = blockIdx.x;
  const float* xr = x + (size_t)row * DIM;
  float s = 0.f, ss = 0.f;
  for (int i = threadIdx.x; i < DIM; i += blockDim.x) {
    float v = xr[i]; s += v; ss += v * v;
  }
  for (int off = 16; off > 0; off >>= 1) {
    s  += __shfl_down(s, off, 32);
    ss += __shfl_down(ss, off, 32);
  }
  int wave = threadIdx.x >> 5, lane = threadIdx.x & 31;
  if (lane == 0) { red_s[wave] = s; red_ss[wave] = ss; }
  __syncthreads();
  if (threadIdx.x == 0) {
    float ts = 0.f, tss = 0.f;
    for (int i = 0; i < (int)(blockDim.x >> 5); i++) { ts += red_s[i]; tss += red_ss[i]; }
    red_s[0] = ts; red_ss[0] = tss;
  }
  __syncthreads();
  float mu   = red_s[0] * (1.0f / DIM);
  float var  = red_ss[0] * (1.0f / DIM) - mu * mu;
  float rstd = rsqrtf(var + 1e-5f);
  for (int i = threadIdx.x; i < DIM; i += blockDim.x) {
    float v = (xr[i] - mu) * rstd * w[i] + b[i];
    out[(size_t)row * DIM + i] = f2bf(v);
  }
}

// ---------------------------------------------------------------- WMMA GEMM (register blocked)
// C[m,n] = sum_k A[m,k] * W[n,k]   (A: MxK bf16 row-major, W: NxK bf16 row-major)
// Wave tile: MT x NT 16x16 tiles (MT*16 rows x NT*16 cols). 8 waves stacked in M.
// Block tile: (8*MT*16) x (NT*16).  grid = (N/(NT*16), M/(8*MT*16))
// mode 0: store bf16          -> outb
// mode 1: store f32           -> outf
// mode 2: softplus(v+bias[n]) -> outf
// mode 3: v + resid[m,n]      -> outf
template <int MT, int NT>
__global__ __launch_bounds__(256)
void gemm_bf16_wmma(const unsigned short* __restrict__ A,
                    const unsigned short* __restrict__ W,
                    int M, int N, int K, int mode,
                    const float* __restrict__ bias,
                    const float* __restrict__ resid,
                    unsigned short* __restrict__ outb,
                    float* __restrict__ outf) {
  int wave = threadIdx.x >> 5;
  int lane = threadIdx.x & 31;
  int g    = lane >> 4;                 // lane half-group
  int l16  = lane & 15;

  int tmBase = (blockIdx.y * 8 + wave) * MT;   // first 16-row tile index
  int tnBase = blockIdx.x * NT;                // first 16-col tile index

  const unsigned short* Arow[MT];
  const unsigned short* Wrow[NT];
  #pragma unroll
  for (int i = 0; i < MT; i++)
    Arow[i] = A + (size_t)((tmBase + i) * 16 + l16) * K;
  #pragma unroll
  for (int j = 0; j < NT; j++)
    Wrow[j] = W + (size_t)((tnBase + j) * 16 + l16) * K;

  v8f acc[MT][NT];
  #pragma unroll
  for (int i = 0; i < MT; i++)
    #pragma unroll
    for (int j = 0; j < NT; j++)
      acc[i][j] = (v8f){};

  FragAB fa[MT], fb[NT];
  for (int kk = 0; kk < K; kk += 32) {
    // A 16x32 bf16 layout: lanes 0-15 K=0..7 / 16..23, lanes 16-31 K=8..15 / 24..31
    #pragma unroll
    for (int i = 0; i < MT; i++) {
      fa[i].q[0] = *reinterpret_cast<const u32x4*>(Arow[i] + kk + g * 8);
      fa[i].q[1] = *reinterpret_cast<const u32x4*>(Arow[i] + kk + g * 8 + 16);
    }
    // B 32x16 bf16 layout: lanes 0-15 K=0..7 / 16..23, lanes 16-31 K=8..15 / 24..31 (col n contiguous in W row)
    #pragma unroll
    for (int j = 0; j < NT; j++) {
      fb[j].q[0] = *reinterpret_cast<const u32x4*>(Wrow[j] + kk + g * 8);
      fb[j].q[1] = *reinterpret_cast<const u32x4*>(Wrow[j] + kk + g * 8 + 16);
    }
    #pragma unroll
    for (int i = 0; i < MT; i++)
      #pragma unroll
      for (int j = 0; j < NT; j++)
        acc[i][j] = __builtin_amdgcn_wmma_f32_16x16x32_bf16(
            false, fa[i].v, false, fb[j].v, (short)0, acc[i][j], false, false);
  }

  #pragma unroll
  for (int i = 0; i < MT; i++) {
    #pragma unroll
    for (int j = 0; j < NT; j++) {
      int nd = (tnBase + j) * 16 + l16;
      #pragma unroll
      for (int e = 0; e < 8; e++) {
        int md = (tmBase + i) * 16 + g * 8 + e;   // D: VGPR e -> M=e (+8 for upper lanes)
        size_t o = (size_t)md * N + nd;
        float v = acc[i][j][e];
        if (mode == 0) {
          outb[o] = f2bf(v);
        } else if (mode == 1) {
          outf[o] = v;
        } else if (mode == 2) {
          float t = v + bias[nd];
          outf[o] = (t > 20.f) ? t : log1pf(__expf(t));
        } else {
          outf[o] = v + resid[o];
        }
      }
    }
  }
}

// ---------------------------------------------------------------- causal dw conv + SiLU
__global__ void conv_silu_kernel(const unsigned short* __restrict__ xz,  // [NROWS, 2*D_INNER] bf16
                                 const float* __restrict__ cw,           // [D_INNER, D_CONV]
                                 const float* __restrict__ cb,           // [D_INNER]
                                 unsigned short* __restrict__ u) {       // [NROWS, D_INNER] bf16
  int idx = blockIdx.x * blockDim.x + threadIdx.x;
  if (idx >= NROWS * D_INNER) return;
  int m = idx / D_INNER;
  int d = idx - m * D_INNER;
  int b = m >> 11;            // / SEQ
  int l = m & (SEQ - 1);
  float acc = cb[d];
  #pragma unroll
  for (int t = 0; t < D_CONV; t++) {
    int ls = l - (D_CONV - 1) + t;
    if (ls >= 0) {
      float xv = bf2f(xz[(size_t)((b << 11) + ls) * (2 * D_INNER) + d]);
      acc += xv * cw[d * D_CONV + t];
    }
  }
  float sg = 1.f / (1.f + __expf(-acc));
  u[idx] = f2bf(acc * sg);
}

// ---------------------------------------------------------------- dt slice -> bf16
__global__ void slice_dt_kernel(const float* __restrict__ dbc,      // [NROWS, 96]
                                unsigned short* __restrict__ dtb) { // [NROWS, 64]
  int idx = blockIdx.x * blockDim.x + threadIdx.x;
  if (idx >= NROWS * DT_RANK) return;
  int m = idx / DT_RANK;
  int r = idx - m * DT_RANK;
  dtb[idx] = f2bf(dbc[(size_t)m * DBC_W + r]);
}

// ---------------------------------------------------------------- selective scan
// One lane per (b, d, state). 16-lane groups reduce y via shfl_xor; next-step
// global loads are software-pipelined ahead of the recurrence math.
__global__ __launch_bounds__(256)
void selective_scan_kernel(const unsigned short* __restrict__ u,     // [NROWS, D_INNER] bf16
                           const float* __restrict__ delta,          // [NROWS, D_INNER]
                           const float* __restrict__ dbc,            // [NROWS, 96] (B@64, C@80)
                           const unsigned short* __restrict__ xz,    // [NROWS, 2*D_INNER] bf16
                           const float* __restrict__ A_log,          // [D_INNER, D_STATE]
                           const float* __restrict__ Dp,             // [D_INNER]
                           unsigned short* __restrict__ y) {         // [NROWS, D_INNER] bf16
  int tid = blockIdx.x * blockDim.x + threadIdx.x;   // BATCH*D_INNER*D_STATE = 65536
  int s  = tid & (D_STATE - 1);
  int rd = tid >> 4;                 // b*D_INNER + d
  int d  = rd & (D_INNER - 1);
  int b  = rd >> 11;

  float Av  = -__expf(A_log[d * D_STATE + s]);
  float dpd = Dp[d];
  float h   = 0.f;
  int rowBase = b << 11;             // b*SEQ

  // prime the pipeline: loads for step 0
  size_t r0 = (size_t)rowBase;
  float ut  = bf2f(u[r0 * D_INNER + d]);
  float dtv = delta[r0 * D_INNER + d];
  float zt  = bf2f(xz[r0 * (2 * D_INNER) + D_INNER + d]);
  float Bv  = dbc[r0 * DBC_W + DT_RANK + s];
  float Cv  = dbc[r0 * DBC_W + DT_RANK + D_STATE + s];

  for (int l = 0; l < SEQ; l++) {
    // prefetch step l+1 while step l computes
    float ut1 = 0.f, dt1 = 0.f, zt1 = 0.f, Bv1 = 0.f, Cv1 = 0.f;
    if (l + 1 < SEQ) {
      size_t r1 = (size_t)(rowBase + l + 1);
      ut1 = bf2f(u[r1 * D_INNER + d]);
      dt1 = delta[r1 * D_INNER + d];
      zt1 = bf2f(xz[r1 * (2 * D_INNER) + D_INNER + d]);
      Bv1 = dbc[r1 * DBC_W + DT_RANK + s];
      Cv1 = dbc[r1 * DBC_W + DT_RANK + D_STATE + s];
    }

    h = __expf(dtv * Av) * h + (dtv * ut) * Bv;
    float yp = h * Cv;
    yp += __shfl_xor(yp, 1);
    yp += __shfl_xor(yp, 2);
    yp += __shfl_xor(yp, 4);
    yp += __shfl_xor(yp, 8);
    if (s == 0) {
      float yt = yp + ut * dpd;
      float sg = 1.f / (1.f + __expf(-zt));
      yt *= zt * sg;
      y[(size_t)(rowBase + l) * D_INNER + d] = f2bf(yt);
    }

    ut = ut1; dtv = dt1; zt = zt1; Bv = Bv1; Cv = Cv1;
  }
}

// ---------------------------------------------------------------- launcher
extern "C" void kernel_launch(void* const* d_in, const int* in_sizes, int n_in,
                              void* d_out, int out_size, void* d_ws, size_t ws_size,
                              hipStream_t stream) {
  const float* x     = (const float*)d_in[0];
  const float* ln_w  = (const float*)d_in[1];
  const float* ln_b  = (const float*)d_in[2];
  const float* in_w  = (const float*)d_in[3];
  const float* cw    = (const float*)d_in[4];
  const float* cb    = (const float*)d_in[5];
  const float* xp_w  = (const float*)d_in[6];
  const float* dt_w  = (const float*)d_in[7];
  const float* dt_b  = (const float*)d_in[8];
  const float* A_log = (const float*)d_in[9];
  const float* Dp    = (const float*)d_in[10];
  const float* out_w = (const float*)d_in[11];
  float* out = (float*)d_out;

  char* ws = (char*)d_ws;
  size_t off = 0;
  auto alloc = [&](size_t bytes) -> char* {
    char* p = ws + off;
    off += (bytes + 255) & ~((size_t)255);
    return p;
  };

  unsigned short* xn     = (unsigned short*)alloc((size_t)NROWS * DIM * 2);
  unsigned short* Wb_in  = (unsigned short*)alloc((size_t)2 * D_INNER * DIM * 2);
  unsigned short* Wb_xp  = (unsigned short*)alloc((size_t)DBC_W * D_INNER * 2);
  unsigned short* Wb_dt  = (unsigned short*)alloc((size_t)D_INNER * DT_RANK * 2);
  unsigned short* Wb_out = (unsigned short*)alloc((size_t)DIM * D_INNER * 2);
  unsigned short* xzb    = (unsigned short*)alloc((size_t)NROWS * 2 * D_INNER * 2);
  unsigned short* ub     = (unsigned short*)alloc((size_t)NROWS * D_INNER * 2);
  float*          dbcf   = (float*)alloc((size_t)NROWS * DBC_W * 4);
  unsigned short* dtb    = (unsigned short*)alloc((size_t)NROWS * DT_RANK * 2);
  float*          deltaf = (float*)alloc((size_t)NROWS * D_INNER * 4);
  unsigned short* yb     = (unsigned short*)alloc((size_t)NROWS * D_INNER * 2);
  (void)ws_size; (void)in_sizes; (void)n_in; (void)out_size;

  const int T = 256;
  // weight conversions fp32 -> bf16
  cvt_f32_to_bf16<<<(2 * D_INNER * DIM + T - 1) / T, T, 0, stream>>>(in_w,  Wb_in,  2 * D_INNER * DIM);
  cvt_f32_to_bf16<<<(DBC_W * D_INNER + T - 1) / T, T, 0, stream>>>(xp_w,  Wb_xp,  DBC_W * D_INNER);
  cvt_f32_to_bf16<<<(D_INNER * DT_RANK + T - 1) / T, T, 0, stream>>>(dt_w,  Wb_dt,  D_INNER * DT_RANK);
  cvt_f32_to_bf16<<<(DIM * D_INNER + T - 1) / T, T, 0, stream>>>(out_w, Wb_out, DIM * D_INNER);

  // layernorm -> xn (bf16)
  layernorm_bf16_kernel<<<NROWS, 256, 0, stream>>>(x, ln_w, ln_b, xn);

  // in_proj: xz = xn @ in_w^T  (4096 x 4096, K=1024) -> bf16
  gemm_bf16_wmma<2, 4><<<dim3((2 * D_INNER) / 64, NROWS / 256), 256, 0, stream>>>(
      xn, Wb_in, NROWS, 2 * D_INNER, DIM, 0, nullptr, nullptr, xzb, nullptr);

  // causal depthwise conv + SiLU -> u (bf16)
  conv_silu_kernel<<<(NROWS * D_INNER + T - 1) / T, T, 0, stream>>>(xzb, cw, cb, ub);

  // x_proj: dbc = u @ xp_w^T  (4096 x 96, K=2048) -> f32   (N = 2 x 48-col tiles)
  gemm_bf16_wmma<2, 3><<<dim3(DBC_W / 48, NROWS / 256), 256, 0, stream>>>(
      ub, Wb_xp, NROWS, DBC_W, D_INNER, 1, nullptr, nullptr, nullptr, dbcf);

  // slice dt columns -> bf16
  slice_dt_kernel<<<(NROWS * DT_RANK + T - 1) / T, T, 0, stream>>>(dbcf, dtb);

  // dt_proj + softplus: delta = softplus(dt @ dt_w^T + dt_b)  (4096 x 2048, K=64) -> f32
  gemm_bf16_wmma<2, 4><<<dim3(D_INNER / 64, NROWS / 256), 256, 0, stream>>>(
      dtb, Wb_dt, NROWS, D_INNER, DT_RANK, 2, dt_b, nullptr, nullptr, deltaf);

  // selective scan (+ D*u + z-gating) -> y (bf16)
  selective_scan_kernel<<<(BATCH * D_INNER * D_STATE) / T, T, 0, stream>>>(
      ub, deltaf, dbcf, xzb, A_log, Dp, yb);

  // out_proj + residual: out = x + y @ out_w^T  (4096 x 1024, K=2048) -> f32
  gemm_bf16_wmma<2, 4><<<dim3(DIM / 64, NROWS / 256), 256, 0, stream>>>(
      yb, Wb_out, NROWS, DIM, D_INNER, 3, nullptr, x, nullptr, out);
}
